// TemporalAttention_15418932593215
// MI455X (gfx1250) — matrix operand order repaired and checked by reading, not compiled, not run on hardware
//
#include <hip/hip_runtime.h>
#include <hip/hip_bf16.h>
#include <cstdint>
#include <cstddef>

// ---------------------------------------------------------------------------
// TemporalAttention on MI455X (gfx1250), wave32 + v_wmma_f32_16x16x32_bf16.
//
//   Qs[i,:] = (X@Wq^T + bq) * invsqrtD * decay[i]          (bf16)
//   K       =  X@Wk^T + bk                                  (bf16, row major)
//   Vt[d,j] = (X@Wv^T + bv)[j,d]                            (bf16, transposed)
//   Z[j]    = sum_i rowmask[i] * exp(S[i,j])                (atomic f32 add)
//   out     = (rowmask[i] * exp(S[i,j]) / Z[j]) @ V         (S recomputed)
// Max-subtraction dropped (softmax shift-invariant; scores are O(1) here).
//
// Round-2 upgrades: register-resident Q/K fragments in the S loops,
// X-fragment reuse in the QKV GEMM, async global->LDS staging (double
// buffered) of the shared Q tile in the column-sum kernel.
// Round-3 fix: async-to-LDS builtin takes (v4i* g, v4i* lds, imm, imm).
// ---------------------------------------------------------------------------

#define NN 8192
#define DD 512
#define INV_SQRT_D 0.04419417382415922f

#ifndef __has_builtin
#define __has_builtin(x) 0
#endif

#if __has_builtin(__builtin_amdgcn_global_load_async_to_lds_b128)
#define HAVE_ASYNC_LDS 1
#else
#define HAVE_ASYNC_LDS 0
#endif

typedef __bf16 bf16_t;
typedef bf16_t v16bf __attribute__((ext_vector_type(16)));
typedef float  v8f   __attribute__((ext_vector_type(8)));
typedef int    v4i   __attribute__((ext_vector_type(4)));

#if HAVE_ASYNC_LDS
typedef __attribute__((address_space(1))) v4i* gptr_v4i;
typedef __attribute__((address_space(3))) v4i* lptr_v4i;
#endif

union FragCvt { v16bf v; float4 f[2]; };

__device__ __forceinline__ unsigned short f2bf(float x) {
    unsigned int u = __float_as_uint(x);
    u += 0x7FFFu + ((u >> 16) & 1u);          // round to nearest even
    return (unsigned short)(u >> 16);
}

// A-matrix 16x32 bf16 frag: lane<16 -> row m=lane, K {0..7}+{16..23};
// lane>=16 -> row m=lane-16, K {8..15}+{24..31}. Row-major source.
__device__ __forceinline__ v16bf load_a_frag(const unsigned short* A, size_t ld,
                                             int i0, int k0, int lane) {
    int half = lane >> 4;
    int m    = lane & 15;
    const unsigned short* p = A + (size_t)(i0 + m) * ld + (size_t)(k0 + half * 8);
    FragCvt fr;
    fr.f[0] = *reinterpret_cast<const float4*>(p);
    fr.f[1] = *reinterpret_cast<const float4*>(p + 16);
    return fr.v;
}

// B-matrix 32x16 bf16 frag with B[k][n] = R[n0+n][k] (column n contiguous in R):
// lane<16 -> col n=lane, K {0..7}+{16..23}; lane>=16 -> col n=lane-16, K {8..15}+{24..31}.
__device__ __forceinline__ v16bf load_b_frag(const unsigned short* R, size_t ld,
                                             int n0, int k0, int lane) {
    int n  = lane & 15;
    int kh = (lane >> 4) * 8;
    const unsigned short* p = R + (size_t)(n0 + n) * ld + (size_t)(k0 + kh);
    FragCvt fr;
    fr.f[0] = *reinterpret_cast<const float4*>(p);
    fr.f[1] = *reinterpret_cast<const float4*>(p + 16);
    return fr.v;
}

__device__ __forceinline__ v8f wmma_bf16(v16bf a, v16bf b, v8f c) {
    return __builtin_amdgcn_wmma_f32_16x16x32_bf16(false, a, false, b,
                                                   (short)0, c, false, false);
}

// 16B global -> LDS copy, async when the gfx1250 builtin is available.
__device__ __forceinline__ void copy16_g2l(const unsigned short* g,
                                           unsigned short* l) {
#if HAVE_ASYNC_LDS
    gptr_v4i gp = (gptr_v4i)(unsigned long long)(uintptr_t)g;
    lptr_v4i lp = (lptr_v4i)(unsigned int)(uintptr_t)l;   // low 32b = LDS offset
    __builtin_amdgcn_global_load_async_to_lds_b128(gp, lp, 0, 0);
#else
    *reinterpret_cast<float4*>(l) = *reinterpret_cast<const float4*>(g);
#endif
}

__device__ __forceinline__ void wait_async_lds() {
#if HAVE_ASYNC_LDS
#if __has_builtin(__builtin_amdgcn_s_wait_asynccnt)
    __builtin_amdgcn_s_wait_asynccnt(0);
#else
    asm volatile("s_wait_asynccnt 0" ::: "memory");
#endif
#endif
}

// ------------------------------ prep kernels -------------------------------

__global__ void ta_cvt_bf16(const float* __restrict__ src,
                            unsigned short* __restrict__ dst, int n) {
    int i = blockIdx.x * blockDim.x + threadIdx.x;
    int stride = gridDim.x * blockDim.x;
    for (; i < n; i += stride) dst[i] = f2bf(src[i]);
}

__global__ void ta_prep_rows(const float* __restrict__ ts,
                             const float* __restrict__ ct,
                             const float* __restrict__ lam,
                             const unsigned char* __restrict__ mask,
                             float* __restrict__ rowscale,
                             float* __restrict__ rowmask) {
    int i = blockIdx.x * blockDim.x + threadIdx.x;
    if (i < NN) {
        float dec = __expf(-lam[0] * (ct[0] - ts[i]));
        rowscale[i] = dec * INV_SQRT_D;
        rowmask[i]  = mask[i] ? 1.0f : 0.0f;
    }
}

__global__ void ta_zero_f32(float* __restrict__ p, int n) {
    int i = blockIdx.x * blockDim.x + threadIdx.x;
    if (i < n) p[i] = 0.0f;
}

__global__ void ta_recip(const float* __restrict__ z, float* __restrict__ iz) {
    int i = blockIdx.x * blockDim.x + threadIdx.x;
    if (i < NN) {
        float v = z[i];
        iz[i] = (v > 0.0f) ? (1.0f / v) : 0.0f;
    }
}

// ------------------------- QKV projection (WMMA) ---------------------------
// grid = (N/16, 1, 3); block = 256 (8 waves). Wave caches the 16x512 X row
// block in registers (16 frags) and computes 4 d-tiles (64 cols) from it.
__global__ __launch_bounds__(256)
void ta_qkv(const unsigned short* __restrict__ Xb,
            const unsigned short* __restrict__ Wqb,
            const unsigned short* __restrict__ Wkb,
            const unsigned short* __restrict__ Wvb,
            const float* __restrict__ bq, const float* __restrict__ bk,
            const float* __restrict__ bv,
            const float* __restrict__ rowscale,
            unsigned short* __restrict__ Qs,
            unsigned short* __restrict__ Kb,
            unsigned short* __restrict__ Vt) {
    const int lane = threadIdx.x & 31;
    const int wave = threadIdx.x >> 5;
    const int i0 = blockIdx.x * 16;
    const int which = blockIdx.z;

    const unsigned short* W = (which == 0) ? Wqb : (which == 1) ? Wkb : Wvb;
    const float*          B = (which == 0) ? bq  : (which == 1) ? bk  : bv;

    v16bf xf[16];
#pragma unroll
    for (int kk = 0; kk < 16; ++kk)
        xf[kk] = load_a_frag(Xb, DD, i0, kk * 32, lane);

    const int nlocal  = lane & 15;
    const int rowbase = (lane >> 4) * 8;

#pragma unroll 1
    for (int t = 0; t < 4; ++t) {
        const int d0 = wave * 64 + t * 16;
        v8f acc = {};
#pragma unroll
        for (int kk = 0; kk < 16; ++kk) {
            v16bf b = load_b_frag(W, DD, d0, kk * 32, lane);
            acc = wmma_bf16(xf[kk], b, acc);
        }

        const int col    = d0 + nlocal;
        const float bias = B[col];

        if (which == 0) {            // Q: scale rows by decay*invsqrtD
#pragma unroll
            for (int r = 0; r < 8; ++r) {
                int row = i0 + rowbase + r;
                Qs[(size_t)row * DD + col] = f2bf((acc[r] + bias) * rowscale[row]);
            }
        } else if (which == 1) {     // K: row-major bf16
#pragma unroll
            for (int r = 0; r < 8; ++r) {
                int row = i0 + rowbase + r;
                Kb[(size_t)row * DD + col] = f2bf(acc[r] + bias);
            }
        } else {                     // V: store transposed Vt[d][j], 16B/lane
            unsigned short tmp[8];
#pragma unroll
            for (int r = 0; r < 8; ++r) tmp[r] = f2bf(acc[r] + bias);
            float4* dst = reinterpret_cast<float4*>(Vt + (size_t)col * NN + i0 + rowbase);
            *dst = *reinterpret_cast<const float4*>(tmp);
        }
    }
}

// ------------------- column denominators Z[j] (WMMA + exp) -----------------
// grid = (N/128, N/128). Wave owns 16 columns: the K fragments for those
// columns are loop-invariant and held in registers. The Q i-tile (shared by
// all 8 waves) is staged global->LDS with async copies, double buffered.
#define SQ_LD 520   // 512 + 8 pad (bank-conflict-free b128 rows)

__global__ __launch_bounds__(256)
void ta_colsum(const unsigned short* __restrict__ Qs,
               const unsigned short* __restrict__ Kb,
               const float* __restrict__ rowmask,
               float* __restrict__ Z) {
    __shared__ __align__(16) unsigned short sQ[2][16][SQ_LD];

    const int tid  = threadIdx.x;
    const int lane = tid & 31;
    const int wave = tid >> 5;
    const int j0 = blockIdx.y * 128 + wave * 16;
    const int ibase = blockIdx.x * 128;
    const int rowbase = (lane >> 4) * 8;

    // loop-invariant K fragments for this wave's 16 columns
    v16bf kf[16];
#pragma unroll
    for (int kk = 0; kk < 16; ++kk)
        kf[kk] = load_b_frag(Kb, DD, j0, kk * 32, lane);

    // stage Q tile 0
#pragma unroll
    for (int q = 0; q < 4; ++q) {
        int c = tid * 4 + q;                  // 1024 chunks of 16B
        int row = c >> 6, col = (c & 63) * 8;
        copy16_g2l(Qs + (size_t)(ibase + row) * DD + col, &sQ[0][row][col]);
    }
    wait_async_lds();
    __syncthreads();

    float cs = 0.0f;
    int buf = 0;
#pragma unroll 1
    for (int it = 0; it < 8; ++it) {
        if (it + 1 < 8) {                     // prefetch next tile async
            const unsigned short* src = Qs + (size_t)(ibase + (it + 1) * 16) * DD;
#pragma unroll
            for (int q = 0; q < 4; ++q) {
                int c = tid * 4 + q;
                int row = c >> 6, col = (c & 63) * 8;
                copy16_g2l(src + (size_t)row * DD + col, &sQ[buf ^ 1][row][col]);
            }
        }

        const unsigned short* sQp = &sQ[buf][0][0];
        v8f acc = {};
#pragma unroll
        for (int kk = 0; kk < 16; ++kk) {
            v16bf a = load_a_frag(sQp, SQ_LD, 0, kk * 32, lane);
            acc = wmma_bf16(a, kf[kk], acc);
        }
        const int i0 = ibase + it * 16;
#pragma unroll
        for (int r = 0; r < 8; ++r)
            cs += rowmask[i0 + rowbase + r] * __expf(acc[r]);

        wait_async_lds();
        __syncthreads();
        buf ^= 1;
    }

    cs += __shfl_xor(cs, 16);
    if (lane < 16) atomicAdd(&Z[j0 + lane], cs);
}

// --------------- fused S recompute -> A (via LDS) -> A@V (WMMA) ------------
// grid = N/16; block = 256 (8 waves). Workgroup owns 16 output rows; the Q
// row-block fragments are register-resident for the whole kernel. Loop over
// j in chunks of 128: each wave builds one 16x16 exp tile -> LDS; all waves
// consume the 16x128 block for their own 64 output columns.
__global__ __launch_bounds__(256)
void ta_out(const unsigned short* __restrict__ Qs,
            const unsigned short* __restrict__ Kb,
            const unsigned short* __restrict__ Vt,
            const float* __restrict__ rowmask,
            const float* __restrict__ invZ,
            float* __restrict__ out) {
    __shared__ __align__(16) unsigned short sA[16][136];  // 16 x 128 (+pad)

    const int lane = threadIdx.x & 31;
    const int wave = threadIdx.x >> 5;
    const int i0 = blockIdx.x * 16;
    const int rowbase = (lane >> 4) * 8;
    const int nlocal = lane & 15;

    float rm[8];
#pragma unroll
    for (int r = 0; r < 8; ++r) rm[r] = rowmask[i0 + rowbase + r];

    // register-resident Q fragments for this row block (16 x 512)
    v16bf qf[16];
#pragma unroll
    for (int kk = 0; kk < 16; ++kk)
        qf[kk] = load_a_frag(Qs, DD, i0, kk * 32, lane);

    v8f accO[4];
#pragma unroll
    for (int t = 0; t < 4; ++t) accO[t] = (v8f){};

#pragma unroll 1
    for (int jc = 0; jc < NN; jc += 128) {
        // ---- phase 1: this wave's 16x16 score tile at columns jc+wave*16 ----
        const int j0 = jc + wave * 16;
        v8f s = {};
#pragma unroll
        for (int kk = 0; kk < 16; ++kk) {
            v16bf b = load_b_frag(Kb, DD, j0, kk * 32, lane);
            s = wmma_bf16(qf[kk], b, s);
        }
        const float iz = invZ[j0 + nlocal];
        const int colLoc = wave * 16 + nlocal;
#pragma unroll
        for (int r = 0; r < 8; ++r)
            sA[rowbase + r][colLoc] = f2bf(rm[r] * __expf(s[r]) * iz);

        __syncthreads();

        // ---- phase 2: A(16x128, LDS) @ V(128 x 64) for this wave's d range ----
        const unsigned short* sAp = &sA[0][0];
#pragma unroll
        for (int t = 0; t < 4; ++t) {
            const int d0 = wave * 64 + t * 16;
#pragma unroll
            for (int jj = 0; jj < 4; ++jj) {
                v16bf a = load_a_frag(sAp, 136, 0, jj * 32, lane);
                v16bf b = load_b_frag(Vt, NN, d0, jc + jj * 32, lane);
                accO[t] = wmma_bf16(a, b, accO[t]);
            }
        }
        __syncthreads();
    }

#pragma unroll
    for (int t = 0; t < 4; ++t) {
        const int col = wave * 64 + t * 16 + nlocal;
#pragma unroll
        for (int r = 0; r < 8; ++r)
            out[(size_t)(i0 + rowbase + r) * DD + col] = accO[t][r];
    }
}

// ------------------------------ launcher -----------------------------------

extern "C" void kernel_launch(void* const* d_in, const int* in_sizes, int n_in,
                              void* d_out, int out_size, void* d_ws, size_t ws_size,
                              hipStream_t stream) {
    const float*         features = (const float*)d_in[0];
    const float*         ts       = (const float*)d_in[1];
    const float*         ct       = (const float*)d_in[2];
    const unsigned char* mask     = (const unsigned char*)d_in[3];
    const float*         lam      = (const float*)d_in[4];
    const float*         Wq       = (const float*)d_in[5];
    const float*         bq       = (const float*)d_in[6];
    const float*         Wk       = (const float*)d_in[7];
    const float*         bk       = (const float*)d_in[8];
    const float*         Wv       = (const float*)d_in[9];
    const float*         bv       = (const float*)d_in[10];
    float*               out      = (float*)d_out;

    char* ws = (char*)d_ws;
    const size_t szND = (size_t)NN * DD * sizeof(unsigned short);  // 8 MB
    const size_t szDD = (size_t)DD * DD * sizeof(unsigned short);  // 512 KB
    const size_t szN  = (size_t)NN * sizeof(float);                // 32 KB

    unsigned short* Xb  = (unsigned short*)(ws);
    unsigned short* Wqb = (unsigned short*)(ws + szND);
    unsigned short* Wkb = (unsigned short*)(ws + szND + szDD);
    unsigned short* Wvb = (unsigned short*)(ws + szND + 2 * szDD);
    unsigned short* Qs  = (unsigned short*)(ws + szND + 3 * szDD);
    unsigned short* Kb  = (unsigned short*)(ws + 2 * szND + 3 * szDD);
    unsigned short* Vt  = (unsigned short*)(ws + 3 * szND + 3 * szDD);
    float* rowscale = (float*)(ws + 4 * szND + 3 * szDD);
    float* rowmask  = (float*)(ws + 4 * szND + 3 * szDD + szN);
    float* Z        = (float*)(ws + 4 * szND + 3 * szDD + 2 * szN);
    float* invZ     = (float*)(ws + 4 * szND + 3 * szDD + 3 * szN);

    // precision conversions + row scales
    ta_cvt_bf16<<<2048, 256, 0, stream>>>(features, Xb, NN * DD);
    ta_cvt_bf16<<<1024, 256, 0, stream>>>(Wq, Wqb, DD * DD);
    ta_cvt_bf16<<<1024, 256, 0, stream>>>(Wk, Wkb, DD * DD);
    ta_cvt_bf16<<<1024, 256, 0, stream>>>(Wv, Wvb, DD * DD);
    ta_prep_rows<<<NN / 256, 256, 0, stream>>>(ts, ct, lam, mask, rowscale, rowmask);
    ta_zero_f32<<<NN / 256, 256, 0, stream>>>(Z, NN);

    // QKV projections
    ta_qkv<<<dim3(NN / 16, 1, 3), 256, 0, stream>>>(
        Xb, Wqb, Wkb, Wvb, bq, bk, bv, rowscale, Qs, Kb, Vt);

    // column softmax denominators
    ta_colsum<<<dim3(NN / 128, NN / 128), 256, 0, stream>>>(Qs, Kb, rowmask, Z);
    ta_recip<<<NN / 256, 256, 0, stream>>>(Z, invZ);

    // fused score recompute + normalize + A@V
    ta_out<<<NN / 16, 256, 0, stream>>>(Qs, Kb, Vt, rowmask, invZ, out);

    (void)in_sizes; (void)n_in; (void)out_size; (void)ws_size;
}
